// LRU_56229711839291
// MI455X (gfx1250) — matrix operand order repaired and checked by reading, not compiled, not run on hardware
//
#include <hip/hip_runtime.h>
#include <cstdint>
#include <cstddef>

#define D_MODEL 1024
#define NSSM    256
#define SEQ_B   4
#define SEQ_L   4096
#define CHUNK   64
#define NCHUNK  (SEQ_L / CHUNK)   // 64
#define K1      D_MODEL           // GEMM1 K = 1024
#define N1      (2 * NSSM)        // stacked re||im = 512
#define MROWS   (SEQ_B * SEQ_L)   // 16384
#define MT      64                // rows per block (both GEMMs)

typedef __bf16 bf16;
typedef __attribute__((ext_vector_type(16))) __bf16 v16bf;
typedef __attribute__((ext_vector_type(8)))  __bf16 v8bf;
typedef __attribute__((ext_vector_type(4)))  __bf16 v4bf;
typedef __attribute__((ext_vector_type(8)))  float  v8f;

static __device__ inline bf16 f2bf(float f) {
  unsigned u = __builtin_bit_cast(unsigned, f);
  unsigned r = (u + 0x7FFFu + ((u >> 16) & 1u)) >> 16;  // round-to-nearest-even
  return __builtin_bit_cast(bf16, (unsigned short)r);
}

union FragU { v16bf v; v8bf h[2]; };

// A fragment (16x32 MxK, bf16) from LDS.
// ISA layout: lane<16 -> M=lane, VGPR0-3: K=half*8+[0..7], VGPR4-7: K=16+half*8+[0..7]
static __device__ inline v16bf load_frag_a_lds(const bf16* lds, int stride, int k0, int lane) {
  int m = lane & 15, half = lane >> 4;
  const bf16* p = lds + m * stride + k0 + half * 8;
  FragU u;
  u.h[0] = *(const v8bf*)(p);        // ds_load_b128
  u.h[1] = *(const v8bf*)(p + 16);   // ds_load_b128
  return u.v;
}

// B fragment (32x16 KxN, bf16) from global; g points at (row = ntile*16, col = k0)
// ISA layout: lane holds N=lane&15, VGPR0-3: K=half*8+[0..7], VGPR4-7: K=16+half*8+[0..7]
static __device__ inline v16bf load_frag_b_glb(const bf16* __restrict__ g, int ldk, int lane) {
  int n = lane & 15, half = lane >> 4;
  const bf16* p = g + n * ldk + half * 8;
  FragU u;
  u.h[0] = *(const v8bf*)(p);        // global_load_b128
  u.h[1] = *(const v8bf*)(p + 16);   // global_load_b128
  return u.v;
}

// ---------------- parameter kernels ----------------

__global__ void setup_kernel(const float* __restrict__ nu_log, const float* __restrict__ theta_log,
                             float* __restrict__ lam_re, float* __restrict__ lam_im,
                             float* __restrict__ lp_re, float* __restrict__ lp_im) {
  int n = threadIdx.x;
  if (n < NSSM) {
    float nu = expf(nu_log[n]);
    float r  = expf(-nu);
    float th = expf(theta_log[n]);
    lam_re[n] = cosf(th) * r;
    lam_im[n] = sinf(th) * r;
    float rC  = expf(-(float)CHUNK * nu);   // r^CHUNK
    float thC = (float)CHUNK * th;
    lp_re[n] = cosf(thC) * rC;
    lp_im[n] = sinf(thC) * rC;
  }
}

// G[512,1024] = [gamma*B_re ; gamma*B_im] bf16 ; Cp[1024,512] = [C_re ; -C_im] bf16
__global__ void pack_kernel(const float* __restrict__ nu_log,
                            const float* __restrict__ B_re, const float* __restrict__ B_im,
                            const float* __restrict__ C_re, const float* __restrict__ C_im,
                            bf16* __restrict__ G, bf16* __restrict__ Cp) {
  int idx = blockIdx.x * 256 + threadIdx.x;     // 4096*256 = 1,048,576 total
  const int GN = N1 * D_MODEL;
  if (idx < GN) {
    int n = idx >> 10, d = idx & 1023;
    int nn = n & (NSSM - 1);
    float r = expf(-expf(nu_log[nn]));
    float gamma = sqrtf(fmaxf(0.0f, 1.0f - r * r));
    float v = (n < NSSM ? B_re[nn * D_MODEL + d] : B_im[nn * D_MODEL + d]) * gamma;
    G[idx] = f2bf(v);
  } else {
    int j = idx - GN;
    int d = j >> 9, c = j & 511;
    float v = (c < NSSM) ? C_re[d * NSSM + c] : -C_im[d * NSSM + (c - NSSM)];
    Cp[j] = f2bf(v);
  }
}

// ---------------- GEMM1: Bu[16384,512] = x[16384,1024] @ G[512,1024]^T ----------------
// Block: 64 rows staged in LDS; 8 waves; wave = 4 m-tiles x 4 n-tiles (16 accs).

__global__ void __launch_bounds__(256) gemm1_bu(const float* __restrict__ X,
                                                const bf16* __restrict__ G,
                                                float* __restrict__ Bu) {
  __shared__ bf16 As[MT * 1032];
  const int STR = 1032;                 // padded, rows stay 16B aligned (2064B)
  int t = threadIdx.x;
  int mbase = blockIdx.x * MT;

  // stage 64x1024 f32 tile -> bf16 LDS (float4 loads, b64 bf16 stores)
  const float4* Xs = (const float4*)(X + (size_t)mbase * K1);
  for (int j = 0; j < 64; ++j) {
    int idx = t + j * 256;              // 16384 float4
    int row = idx >> 8, c4 = idx & 255;
    float4 v = Xs[idx];
    v4bf o; o[0] = f2bf(v.x); o[1] = f2bf(v.y); o[2] = f2bf(v.z); o[3] = f2bf(v.w);
    *(v4bf*)(As + row * STR + c4 * 4) = o;
  }
  __syncthreads();

  int lane = t & 31, wave = t >> 5;     // 8 waves: wave w -> n-tiles w*4..w*4+3
  v8f acc[4][4] = {};                   // [mi][q]
  for (int kt = 0; kt < K1 / 32; ++kt) {
    int k0 = kt * 32;
    v16bf a[4], b[4];
#pragma unroll
    for (int mi = 0; mi < 4; ++mi)
      a[mi] = load_frag_a_lds(As + mi * 16 * STR, STR, k0, lane);
#pragma unroll
    for (int q = 0; q < 4; ++q)
      b[q] = load_frag_b_glb(G + (wave * 4 + q) * 16 * K1 + k0, K1, lane);
#pragma unroll
    for (int mi = 0; mi < 4; ++mi)
#pragma unroll
      for (int q = 0; q < 4; ++q)
        acc[mi][q] = __builtin_amdgcn_wmma_f32_16x16x32_bf16(false, a[mi], false, b[q],
                                                             (short)0, acc[mi][q], false, false);
  }

  int half = lane >> 4, col = lane & 15;
#pragma unroll
  for (int mi = 0; mi < 4; ++mi)
#pragma unroll
    for (int q = 0; q < 4; ++q) {
      int nb = (wave * 4 + q) * 16 + col;
#pragma unroll
      for (int i = 0; i < 8; ++i) {
        int m = mbase + mi * 16 + i + half * 8;
        Bu[m * N1 + nb] = acc[mi][q][i];
      }
    }
}

// ---------------- chunked complex scan over L ----------------

__global__ void __launch_bounds__(256) scanA_kernel(const float* __restrict__ Bu,
                                                    const float* __restrict__ lam_re,
                                                    const float* __restrict__ lam_im,
                                                    float* __restrict__ S) {
  int n = threadIdx.x;
  int b = blockIdx.x >> 6;
  int c = blockIdx.x & 63;
  float lr = lam_re[n], li = lam_im[n];
  const float* base = Bu + (size_t)(b * SEQ_L + c * CHUNK) * N1 + n;
  float sr = 0.f, si = 0.f;
  for (int j = 0; j < CHUNK; ++j) {
    float br = base[(size_t)j * N1];
    float bi = base[(size_t)j * N1 + NSSM];
    float nr = fmaf(lr, sr, fmaf(-li, si, br));
    float ni = fmaf(li, sr, fmaf(lr, si, bi));
    sr = nr; si = ni;
  }
  size_t o = 2 * ((size_t)(b * NSSM + n) * NCHUNK + c);
  S[o] = sr; S[o + 1] = si;
}

__global__ void scanMid_kernel(const float* __restrict__ S,
                               const float* __restrict__ lp_re, const float* __restrict__ lp_im,
                               float* __restrict__ P) {
  int idx = blockIdx.x * 256 + threadIdx.x;   // 0..1023 = b*256+n
  int n = idx & (NSSM - 1);
  float lr = lp_re[n], li = lp_im[n];
  float pr = 0.f, pi = 0.f;
  size_t base = (size_t)idx * NCHUNK * 2;
  for (int c = 0; c < NCHUNK; ++c) {
    size_t o = base + 2 * c;
    P[o] = pr; P[o + 1] = pi;
    float sr = S[o], si = S[o + 1];
    float nr = fmaf(lr, pr, fmaf(-li, pi, sr));
    float ni = fmaf(li, pr, fmaf(lr, pi, si));
    pr = nr; pi = ni;
  }
}

__global__ void __launch_bounds__(256) scanB_kernel(const float* __restrict__ Bu,
                                                    const float* __restrict__ lam_re,
                                                    const float* __restrict__ lam_im,
                                                    const float* __restrict__ P,
                                                    bf16* __restrict__ H) {
  int n = threadIdx.x;
  int b = blockIdx.x >> 6;
  int c = blockIdx.x & 63;
  float lr = lam_re[n], li = lam_im[n];
  size_t po = 2 * ((size_t)(b * NSSM + n) * NCHUNK + c);
  float hr = P[po], hi = P[po + 1];
  size_t row0 = (size_t)(b * SEQ_L + c * CHUNK);
  const float* base = Bu + row0 * N1 + n;
  bf16* hb = H + row0 * N1 + n;
  for (int j = 0; j < CHUNK; ++j) {
    float br = base[(size_t)j * N1];
    float bi = base[(size_t)j * N1 + NSSM];
    float nr = fmaf(lr, hr, fmaf(-li, hi, br));
    float ni = fmaf(li, hr, fmaf(lr, hi, bi));
    hr = nr; hi = ni;
    hb[(size_t)j * N1]        = f2bf(hr);
    hb[(size_t)j * N1 + NSSM] = f2bf(hi);
  }
}

// ---------------- GEMM2: out[16384,1024] = H[16384,512] @ Cp[1024,512]^T + D*x --------
// Block: 64 rows x 512 cols; A tile staged via async DMA to LDS (ASYNCcnt path).

__global__ void __launch_bounds__(256) gemm2_out(const bf16* __restrict__ H,
                                                 const bf16* __restrict__ Cp,
                                                 const float* __restrict__ X,
                                                 const float* __restrict__ Dvec,
                                                 float* __restrict__ Out) {
  __shared__ bf16 As[MT * 520];         // single shared object -> LDS offset 0
  const int STR = 520;                  // 1040B row stride, 16B aligned
  int t = threadIdx.x;
  int mblk = blockIdx.x >> 1;
  int nh   = blockIdx.x & 1;            // which half of the 1024 output cols
  int mbase = mblk * MT;

  // async-DMA stage: 64 rows x 1024B = 4096 x b128 transfers, 16 per thread
  const char* gbase = (const char*)(H + (size_t)mbase * N1);
  for (int j = 0; j < 16; ++j) {
    int idx = t + j * 256;
    int row = idx >> 6, cp = idx & 63;
    unsigned lds_off = (unsigned)(row * (STR * 2) + cp * 16);
    uint64_t ga = (uint64_t)(gbase + (size_t)row * 1024 + cp * 16);
    asm volatile("global_load_async_to_lds_b128 %0, %1, off"
                 :: "v"(lds_off), "v"(ga) : "memory");
  }
  asm volatile("s_wait_asynccnt 0x0" ::: "memory");
  __syncthreads();

  int lane = t & 31, wave = t >> 5;     // wave w -> n-tiles nh*32 + w*4 .. +3
  v8f acc[4][4] = {};
  for (int kt = 0; kt < N1 / 32; ++kt) {
    int k0 = kt * 32;
    v16bf a[4], b[4];
#pragma unroll
    for (int mi = 0; mi < 4; ++mi)
      a[mi] = load_frag_a_lds(As + mi * 16 * STR, STR, k0, lane);
#pragma unroll
    for (int q = 0; q < 4; ++q) {
      int nt = nh * 32 + wave * 4 + q;
      b[q] = load_frag_b_glb(Cp + nt * 16 * N1 + k0, N1, lane);
    }
#pragma unroll
    for (int mi = 0; mi < 4; ++mi)
#pragma unroll
      for (int q = 0; q < 4; ++q)
        acc[mi][q] = __builtin_amdgcn_wmma_f32_16x16x32_bf16(false, a[mi], false, b[q],
                                                             (short)0, acc[mi][q], false, false);
  }

  int half = lane >> 4, col = lane & 15;
#pragma unroll
  for (int mi = 0; mi < 4; ++mi)
#pragma unroll
    for (int q = 0; q < 4; ++q) {
      int d0 = (nh * 32 + wave * 4 + q) * 16 + col;
      float dv = Dvec[d0];
#pragma unroll
      for (int i = 0; i < 8; ++i) {
        int m = mbase + mi * 16 + i + half * 8;
        size_t o = (size_t)m * D_MODEL + d0;
        Out[o] = acc[mi][q][i] + dv * X[o];
      }
    }
}

// ---------------- launch ----------------

extern "C" void kernel_launch(void* const* d_in, const int* in_sizes, int n_in,
                              void* d_out, int out_size, void* d_ws, size_t ws_size,
                              hipStream_t stream) {
  const float* x      = (const float*)d_in[0];
  const float* nu_log = (const float*)d_in[1];
  const float* th_log = (const float*)d_in[2];
  const float* B_re   = (const float*)d_in[3];
  const float* B_im   = (const float*)d_in[4];
  const float* C_re   = (const float*)d_in[5];
  const float* C_im   = (const float*)d_in[6];
  const float* Dv     = (const float*)d_in[7];
  float* out = (float*)d_out;

  char* w = (char*)d_ws;
  float* lam_re = (float*)(w + 0);
  float* lam_im = (float*)(w + 1024);
  float* lp_re  = (float*)(w + 2048);
  float* lp_im  = (float*)(w + 3072);
  size_t off = 4096;
  bf16*  G    = (bf16*)(w + off);  off += (size_t)N1 * D_MODEL * 2;
  bf16*  Cp   = (bf16*)(w + off);  off += (size_t)D_MODEL * N1 * 2;
  float* Bu   = (float*)(w + off); off += (size_t)MROWS * N1 * 4;
  bf16*  H    = (bf16*)(w + off);  off += (size_t)MROWS * N1 * 2;
  float* Sagg = (float*)(w + off); off += (size_t)SEQ_B * NSSM * NCHUNK * 2 * 4;
  float* Ppre = (float*)(w + off);

  setup_kernel<<<1, 256, 0, stream>>>(nu_log, th_log, lam_re, lam_im, lp_re, lp_im);
  pack_kernel<<<4096, 256, 0, stream>>>(nu_log, B_re, B_im, C_re, C_im, G, Cp);
  gemm1_bu<<<MROWS / MT, 256, 0, stream>>>(x, G, Bu);
  scanA_kernel<<<SEQ_B * NCHUNK, 256, 0, stream>>>(Bu, lam_re, lam_im, Sagg);
  scanMid_kernel<<<4, 256, 0, stream>>>(Sagg, lp_re, lp_im, Ppre);
  scanB_kernel<<<SEQ_B * NCHUNK, 256, 0, stream>>>(Bu, lam_re, lam_im, Ppre, H);
  gemm2_out<<<(MROWS / MT) * 2, 256, 0, stream>>>(H, Cp, x, Dv, out);
}
